// Attention_3109556322645
// MI455X (gfx1250) — compile-verified, matrix operand order checked by reading, not run on hardware
//
#include <hip/hip_runtime.h>

#define Hdim 512
#define H2   1024   // 2H
#define H4   2048   // 4H
#define Bsz  32
#define Ssz  2048

typedef __attribute__((ext_vector_type(16))) __bf16         v16bf;
typedef __attribute__((ext_vector_type(8)))  float          v8f;
typedef __attribute__((ext_vector_type(4)))  float          f4;
typedef __attribute__((ext_vector_type(8)))  unsigned short us8;

// fp32 -> bf16 bits, round-to-nearest-even
__device__ __forceinline__ unsigned short f2bf(float f) {
    unsigned int u = __float_as_uint(f);
    unsigned int r = (u + 0x7FFFu + ((u >> 16) & 1u)) >> 16;
    return (unsigned short)r;
}

// ---------------------------------------------------------------------------
// 1) hp[b,h] = bias[h] + hidden[b, 0:2H] . W[h, 0:2H]      (W1 half)
// ---------------------------------------------------------------------------
__global__ void hp_kernel(const float* __restrict__ hidden,
                          const float* __restrict__ W,
                          const float* __restrict__ bias,
                          float* __restrict__ hp) {
    int idx = blockIdx.x * 256 + threadIdx.x;     // 32*512 = 16384 total
    int b = idx >> 9;
    int h = idx & 511;
    const float* hrow = hidden + (size_t)b * H2;
    const float* wrow = W + (size_t)h * H4;       // first 2H columns = W1
    float sum = bias[h];
    #pragma unroll 4
    for (int f = 0; f < H2; ++f) sum += hrow[f] * wrow[f];
    hp[idx] = sum;
}

// ---------------------------------------------------------------------------
// 2) W2 (= W[:, 2H:4H]) fp32 -> bf16, row-major [H][2H]
// ---------------------------------------------------------------------------
__global__ void w2cvt_kernel(const float* __restrict__ W,
                             unsigned short* __restrict__ w2) {
    int idx = blockIdx.x * 256 + threadIdx.x;     // 512*1024 = 524288 total
    int h = idx >> 10;
    int k = idx & 1023;
    w2[idx] = f2bf(W[(size_t)h * H4 + H2 + k]);
}

// ---------------------------------------------------------------------------
// 3) Main fused GEMM + tanh + v-dot:
//    scores[b,s] = sum_h tanh( enc[b,s,:] . W2[h,:] + hp[b,h] ) * v[h]
//    grid = (32 s-tiles, 32 batches), block = 128 threads (4 waves).
//    Each workgroup: 64 enc rows staged fp32->bf16 in 128KB LDS.
//    Each wave: 16 rows (M=16), loops 32 h-tiles (N=16) x 32 k-steps (K=32).
// ---------------------------------------------------------------------------
__global__ void scores_kernel(const float* __restrict__ enc,
                              const unsigned short* __restrict__ w2,
                              const float* __restrict__ hp,
                              const float* __restrict__ v,
                              float* __restrict__ scores) {
    extern __shared__ unsigned short ldsA[];      // [64][1024] bf16 = 128 KB
    const int stile = blockIdx.x;                 // 0..31
    const int b     = blockIdx.y;                 // 0..31
    const int tid   = threadIdx.x;                // 0..127
    const int lane  = tid & 31;
    const int wave  = tid >> 5;                   // 0..3
    const int sBase = stile * 64;

    // ---- stage enc tile [64 x 1024] fp32 -> bf16 in LDS (coalesced f4) ----
    const float* encTile = enc + ((size_t)b * Ssz + sBase) * H2;
    for (int i = tid; i < (64 * H2) / 4; i += 128) {
        f4 x = *(const f4*)(encTile + (size_t)i * 4);
        int o = i * 4;
        ldsA[o + 0] = f2bf(x[0]);
        ldsA[o + 1] = f2bf(x[1]);
        ldsA[o + 2] = f2bf(x[2]);
        ldsA[o + 3] = f2bf(x[3]);
    }
    __syncthreads();

    // ISA 16-bit fragment layout: lanes 0-15 hold K {0..7, 16..23},
    // lanes 16-31 hold K {8..15, 24..31}; row/col = lane % 16.
    const int halfsel  = lane >> 4;               // 0 or 1
    const int ln       = lane & 15;
    const int rowLocal = wave * 16 + ln;          // A row this lane serves
    const int koffLane = halfsel * 8;

    float part[8];
    #pragma unroll
    for (int j = 0; j < 8; ++j) part[j] = 0.0f;

    for (int ht = 0; ht < Hdim / 16; ++ht) {      // 32 N-tiles
        const int hCol = ht * 16 + ln;            // B column this lane serves
        const unsigned short* w2col = w2 + (size_t)hCol * H2;
        v8f c = {};
        #pragma unroll 4
        for (int k = 0; k < H2 / 32; ++k) {       // 32 K-steps
            const int kb = k * 32 + koffLane;
            us8 a_lo = *(const us8*)(&ldsA[rowLocal * H2 + kb]);
            us8 a_hi = *(const us8*)(&ldsA[rowLocal * H2 + kb + 16]);
            us8 b_lo = *(const us8*)(w2col + kb);
            us8 b_hi = *(const us8*)(w2col + kb + 16);
            union { us8 u[2]; v16bf v; } ua, ub;
            ua.u[0] = a_lo; ua.u[1] = a_hi;
            ub.u[0] = b_lo; ub.u[1] = b_hi;
            c = __builtin_amdgcn_wmma_f32_16x16x32_bf16(
                    false, ua.v, false, ub.v, (short)0, c, false, false);
        }
        // epilogue for this N-tile: tanh + scale by v, accumulate per lane
        float hpv = hp[b * Hdim + hCol];
        float vv  = v[hCol];
        #pragma unroll
        for (int j = 0; j < 8; ++j)
            part[j] += tanhf(c[j] + hpv) * vv;    // C: VGPR j -> M=j (+8 hi half)
    }

    // reduce over the 16 lanes of each half (N dimension)
    #pragma unroll
    for (int j = 0; j < 8; ++j) {
        float x = part[j];
        x += __shfl_xor(x, 1, 32);
        x += __shfl_xor(x, 2, 32);
        x += __shfl_xor(x, 4, 32);
        x += __shfl_xor(x, 8, 32);
        part[j] = x;
    }
    if (ln == 0) {
        int rowM = sBase + wave * 16 + halfsel * 8;   // lanes 0-15: M=j, 16-31: M=j+8
        #pragma unroll
        for (int j = 0; j < 8; ++j)
            scores[(size_t)b * Ssz + rowM + j] = part[j];
    }
}

// ---------------------------------------------------------------------------
// 4) softmax over S per batch
// ---------------------------------------------------------------------------
__global__ void softmax_kernel(const float* __restrict__ scores,
                               float* __restrict__ attn) {
    __shared__ float red[256];
    int b = blockIdx.x, tid = threadIdx.x;
    float vals[8];
    float lmax = -3.0e38f;
    #pragma unroll
    for (int i = 0; i < 8; ++i) {
        vals[i] = scores[(size_t)b * Ssz + tid + i * 256];
        lmax = fmaxf(lmax, vals[i]);
    }
    red[tid] = lmax; __syncthreads();
    for (int s = 128; s > 0; s >>= 1) {
        if (tid < s) red[tid] = fmaxf(red[tid], red[tid + s]);
        __syncthreads();
    }
    float m = red[0]; __syncthreads();
    float lsum = 0.0f;
    #pragma unroll
    for (int i = 0; i < 8; ++i) { vals[i] = __expf(vals[i] - m); lsum += vals[i]; }
    red[tid] = lsum; __syncthreads();
    for (int s = 128; s > 0; s >>= 1) {
        if (tid < s) red[tid] += red[tid + s];
        __syncthreads();
    }
    float inv = 1.0f / red[0];
    #pragma unroll
    for (int i = 0; i < 8; ++i)
        attn[(size_t)b * Ssz + tid + i * 256] = vals[i] * inv;
}

// ---------------------------------------------------------------------------
// 5) context[b,d] = sum_s attn[b,s] * enc[b,s,d]   (memory-bound, coalesced)
//    grid = 128 blocks (4 d-chunks per batch), 256 threads
// ---------------------------------------------------------------------------
__global__ void context_kernel(const float* __restrict__ enc,
                               const float* __restrict__ attn,
                               float* __restrict__ out) {
    __shared__ float at[Ssz];                     // 8 KB
    int blk = blockIdx.x;
    int b = blk >> 2;
    int d = ((blk & 3) << 8) + threadIdx.x;
    for (int s = threadIdx.x; s < Ssz; s += 256)
        at[s] = attn[(size_t)b * Ssz + s];
    __syncthreads();
    float acc = 0.0f;
    const float* base = enc + (size_t)b * Ssz * H2 + d;
    #pragma unroll 4
    for (int s = 0; s < Ssz; ++s)
        acc += at[s] * base[(size_t)s * H2];
    out[b * H2 + d] = acc;
}

// ---------------------------------------------------------------------------
extern "C" void kernel_launch(void* const* d_in, const int* in_sizes, int n_in,
                              void* d_out, int out_size, void* d_ws, size_t ws_size,
                              hipStream_t stream) {
    const float* hidden = (const float*)d_in[0];   // [32, 1024]
    const float* enc    = (const float*)d_in[1];   // [32, 2048, 1024]
    const float* W      = (const float*)d_in[2];   // [512, 2048]
    const float* bias   = (const float*)d_in[3];   // [512]
    const float* v      = (const float*)d_in[4];   // [512]
    float* out = (float*)d_out;                    // [32, 1024]

    char* ws = (char*)d_ws;
    float*          hp     = (float*)(ws);                               // 64 KB
    unsigned short* w2     = (unsigned short*)(ws + 65536);              // 1 MB bf16
    float*          scores = (float*)(ws + 65536 + 1048576);             // 256 KB
    float*          attn   = (float*)(ws + 65536 + 1048576 + 262144);    // 256 KB

    hp_kernel   <<<64,   256, 0, stream>>>(hidden, W, bias, hp);
    w2cvt_kernel<<<2048, 256, 0, stream>>>(W, w2);

    dim3 g(Ssz / 64, Bsz);
    scores_kernel<<<g, 128, 64 * H2 * sizeof(unsigned short), stream>>>(
        enc, w2, hp, v, scores);

    softmax_kernel<<<Bsz, 256, 0, stream>>>(scores, attn);
    context_kernel<<<Bsz * (H2 / 256), 256, 0, stream>>>(enc, attn, out);
}
